// FEFM_48739288875209
// MI455X (gfx1250) — compile-verified, matrix operand order checked by reading, not run on hardware
//
#include <hip/hip_runtime.h>
#include <hip/hip_bf16.h>

// ---------------------------------------------------------------------------
// FEFM block for MI455X (gfx1250).  All large GEMMs run on
// v_wmma_f32_16x16x32_bf16 (bf16 in / f32 accum); FFT2 stays f32 in LDS.
// ---------------------------------------------------------------------------

typedef __attribute__((ext_vector_type(8)))  float  v8f;
typedef __attribute__((ext_vector_type(8)))  __bf16 bf16x8;
typedef __attribute__((ext_vector_type(16))) __bf16 bf16x16;

#define NB   8          // batch
#define NC   256        // MID channels
#define HW   4096       // 64*64
#define NPIX 32768      // NB*HW
#define TOTE (1LL<<23)  // NB*NC*HW elements

// ---------------------------------------------------------------------------
// Generic batched WMMA GEMM:  D[m][n] = [ACCUM]*D + scale*sum_k A[m][k]*B[n][k]
//                                        (+ bias[n] if HAS_BIAS)
// A: [M x K] bf16 row-major, B: [N x K] bf16 row-major (i.e. D = A * B^T).
// Block = 128 threads (4 waves), block tile 128x64, wave tile 64x32
// (4x2 WMMA tiles -> 8 v_wmma per K-step against 12 b128 loads; B fragments
// are reused across the 4 A fragments).
// A-fragment per lane: row m0+lr, k = kk + half*8 + {0..7} and +16 -> two
// b128 loads exactly matching the CDNA5 16-bit A/B VGPR layout.
// ---------------------------------------------------------------------------
template<bool HAS_BIAS, bool ACCUM>
__global__ __launch_bounds__(128)
void gemm_abT_bf16(const __bf16* __restrict__ A, const __bf16* __restrict__ Bm,
                   float* __restrict__ D, const float* __restrict__ bias,
                   int K, int lda, int ldb, int ldd,
                   long long sA, long long sB, long long sD,
                   float scale)
{
    int z = blockIdx.z;
    A  += (long long)z * sA;
    Bm += (long long)z * sB;
    D  += (long long)z * sD;

    const int lane = threadIdx.x & 31;
    const int wv   = threadIdx.x >> 5;              // 0..3
    const int m0   = blockIdx.x * 128 + (wv >> 1) * 64;
    const int n0   = blockIdx.y * 64  + (wv & 1) * 32;
    const int half = lane >> 4;                     // K-half select
    const int lr   = lane & 15;                     // row/col within fragment

    v8f acc[4][2] = {};

    for (int kk = 0; kk < K; kk += 32) {
        bf16x16 bf[2];
#pragma unroll
        for (int j = 0; j < 2; ++j) {
            const __bf16* pb = Bm + (long long)(n0 + j*16 + lr) * ldb + kk + half*8;
            bf16x8 blo = *(const bf16x8*)pb;
            bf16x8 bhi = *(const bf16x8*)(pb + 16);
            bf[j] = __builtin_shufflevector(blo, bhi,
                     0,1,2,3,4,5,6,7,8,9,10,11,12,13,14,15);
        }
        bf16x16 af[4];
#pragma unroll
        for (int i = 0; i < 4; ++i) {
            const __bf16* pa = A + (long long)(m0 + i*16 + lr) * lda + kk + half*8;
            bf16x8 alo = *(const bf16x8*)pa;
            bf16x8 ahi = *(const bf16x8*)(pa + 16);
            af[i] = __builtin_shufflevector(alo, ahi,
                     0,1,2,3,4,5,6,7,8,9,10,11,12,13,14,15);
            __builtin_prefetch(pa + 32, 0, 1);      // global_prefetch next K tile
        }
#pragma unroll
        for (int i = 0; i < 4; ++i)
#pragma unroll
            for (int j = 0; j < 2; ++j)
                acc[i][j] = __builtin_amdgcn_wmma_f32_16x16x32_bf16(
                    false, af[i], false, bf[j], (short)0, acc[i][j], false, false);
    }

    // C/D layout: VGPR r -> M = r + 8*half ; lane%16 -> N.  Branchless epilogue.
#pragma unroll
    for (int j = 0; j < 2; ++j) {
        const int n = n0 + j*16 + lr;
        const float bv = HAS_BIAS ? bias[n] : 0.f;
#pragma unroll
        for (int i = 0; i < 4; ++i)
#pragma unroll
            for (int r = 0; r < 8; ++r) {
                int m = m0 + i*16 + r + half*8;
                long long di = (long long)m * ldd + n;
                float v = scale * acc[i][j][r] + bv;
                if (ACCUM) v += D[di];
                D[di] = v;
            }
    }
}

// ---------------------------------------------------------------------------
// Packing kernels
// ---------------------------------------------------------------------------
__global__ void pack_flat_bf16(const float* __restrict__ s, __bf16* __restrict__ d,
                               long long n)
{
    long long i = (long long)blockIdx.x * 256 + threadIdx.x;
    if (i < n) d[i] = (__bf16)s[i];
}

// NCHW f32 -> pixel-major [b*HW+p][c] bf16
__global__ void pack_nchw_to_pix_bf16(const float* __restrict__ s,
                                      __bf16* __restrict__ d, int C)
{
    long long idx = (long long)blockIdx.x * 256 + threadIdx.x;
    int c = (int)(idx % C);
    long long q = idx / C;
    int p = (int)(q & 4095);
    int b = (int)(q >> 12);
    d[idx] = (__bf16)s[((long long)b * C + c) * HW + p];
}

// pixel-major f32 -> NCHW f32 (final output)
__global__ void to_nchw(const float* __restrict__ pix, float* __restrict__ out)
{
    long long idx = (long long)blockIdx.x * 256 + threadIdx.x; // NCHW flat
    int p = (int)(idx & 4095);
    int c = (int)((idx >> 12) & 255);
    int b = (int)(idx >> 20);
    out[idx] = pix[((long long)b * HW + p) * NC + c];
}

// ---------------------------------------------------------------------------
// Pool + MLP gate
// ---------------------------------------------------------------------------
__global__ void pool_mean(const float* __restrict__ FR, const float* __restrict__ FN,
                          float* __restrict__ pooled)
{
    int b = blockIdx.x, c = threadIdx.x;
    float s = 0.f;
    for (int p = 0; p < HW; ++p) {
        long long i = ((long long)b * HW + p) * NC + c;
        s += FR[i] + FN[i];
    }
    pooled[b * NC + c] = s * (1.f / HW);
}

__global__ __launch_bounds__(512)
void mlp_gate(const float* __restrict__ pooled, const float* __restrict__ w1,
              const float* __restrict__ w2, float* __restrict__ attn2)
{
    __shared__ float po[256];
    __shared__ float h[32];
    __shared__ float a[512];
    int t = threadIdx.x;
    for (int b = 0; b < NB; ++b) {
        if (t < 256) po[t] = pooled[b * 256 + t];
        __syncthreads();
        if (t < 32) {
            float s = 0.f;
            for (int c = 0; c < 256; ++c) s += po[c] * w1[t * 256 + c];
            h[t] = fmaxf(s, 0.f);
        }
        __syncthreads();
        {
            float s = 0.f;
            for (int d = 0; d < 32; ++d) s += h[d] * w2[t * 32 + d];
            a[t] = s;
        }
        __syncthreads();
        if (t < 256) {
            float a0 = a[t], a1 = a[t + 256];
            float m  = fmaxf(a0, a1);
            float e0 = __expf(a0 - m), e1 = __expf(a1 - m);
            float inv = 1.f / (e0 + e1);
            attn2[b * 512 + t]       = e0 * inv;
            attn2[b * 512 + 256 + t] = e1 * inv;
        }
        __syncthreads();
    }
}

// Fw = a0*FR + a1*FN  (pixel-major) + bf16 pack for next GEMMs
__global__ void blend_fw(const float* __restrict__ FR, const float* __restrict__ FN,
                         const float* __restrict__ attn2,
                         float* __restrict__ Fw, __bf16* __restrict__ Fwp)
{
    long long idx = (long long)blockIdx.x * 256 + threadIdx.x;
    int c = (int)(idx & 255);
    int b = (int)(idx >> 20);
    float a0 = attn2[b * 512 + c], a1 = attn2[b * 512 + 256 + c];
    float v  = a0 * FR[idx] + a1 * FN[idx];
    Fw[idx]  = v;
    Fwp[idx] = (__bf16)v;
}

// ---------------------------------------------------------------------------
// Depthwise 3x3 (SAME/zero pad).  Reads pixel-major pre-act, writes NCHW.
// ---------------------------------------------------------------------------
__global__ void dwconv3x3(const float* __restrict__ pre, const float* __restrict__ w,
                          const float* __restrict__ bias, float* __restrict__ out)
{
    long long idx = (long long)blockIdx.x * 256 + threadIdx.x; // NCHW flat
    int p = (int)(idx & 4095);
    int c = (int)((idx >> 12) & 255);
    int b = (int)(idx >> 20);
    int hh = p >> 6, ww = p & 63;
    float s = bias[c];
#pragma unroll
    for (int i = 0; i < 3; ++i)
#pragma unroll
        for (int j = 0; j < 3; ++j) {
            int y = hh + i - 1, x = ww + j - 1;
            if (y >= 0 && y < 64 && x >= 0 && x < 64)
                s += pre[((long long)b * HW + y * 64 + x) * NC + c] * w[c * 9 + i * 3 + j];
        }
    out[idx] = s;
}

// ---------------------------------------------------------------------------
// FFT2 64x64 via two LDS-resident 64-pt DFT passes (f32).  One block per
// (b,c) plane; twiddle LUT + both stage buffers live in LDS (dynamic: 64.5KB,
// comfortably inside the 320KB WGP LDS of CDNA5).
// ---------------------------------------------------------------------------
__global__ __launch_bounds__(256)
void fft2_plane(const float* __restrict__ inr, const float* __restrict__ ini,
                float* __restrict__ outr, float* __restrict__ outi,
                __bf16* __restrict__ obr, __bf16* __restrict__ obi,
                float sign, float scale)
{
    extern __shared__ float sm[];
    float* xr  = sm;
    float* xi  = sm + 4096;
    float* tr  = sm + 8192;
    float* ti  = sm + 12288;
    float* twr = sm + 16384;
    float* twi = sm + 16448;

    long long base = (long long)blockIdx.x * HW;
    int tid = threadIdx.x;

    for (int i = tid; i < HW; i += 256) {
        xr[i] = inr[base + i];
        xi[i] = ini ? ini[base + i] : 0.f;
    }
    if (tid < 64) {
        float ang = sign * 6.2831853071795864769f * (float)tid * (1.f / 64.f);
        twr[tid] = __cosf(ang);
        twi[tid] = __sinf(ang);
    }
    __syncthreads();

    // rows: T[h][k] = sum_w tw[w*k] * X[h][w]
    for (int o = tid; o < HW; o += 256) {
        int h = o >> 6, k = o & 63;
        const float* rr = xr + h * 64;
        const float* ri = xi + h * 64;
        float sr = 0.f, si = 0.f;
        for (int w = 0; w < 64; ++w) {
            int t = (w * k) & 63;
            float cr = twr[t], ci = twi[t];
            float ar = rr[w], ai = ri[w];
            sr += ar * cr - ai * ci;
            si += ar * ci + ai * cr;
        }
        tr[o] = sr; ti[o] = si;
    }
    __syncthreads();

    // cols: Y[j][k] = sum_h tw[h*j] * T[h][k]
    for (int o = tid; o < HW; o += 256) {
        int j = o >> 6, k = o & 63;
        float sr = 0.f, si = 0.f;
        for (int h = 0; h < 64; ++h) {
            int t = (h * j) & 63;
            float cr = twr[t], ci = twi[t];
            float ar = tr[h * 64 + k], ai = ti[h * 64 + k];
            sr += ar * cr - ai * ci;
            si += ar * ci + ai * cr;
        }
        sr *= scale; si *= scale;
        outr[base + o] = sr;
        if (outi) outi[base + o] = si;
        if (obr)  obr[base + o] = (__bf16)sr;
        if (obi)  obi[base + o] = (__bf16)si;
    }
}

// ---------------------------------------------------------------------------
// softmax over last dim of |attn|/alpha ; rows = B*C, cols = C. bf16 output.
// ---------------------------------------------------------------------------
__global__ __launch_bounds__(256)
void attn_softmax(const float* __restrict__ re, const float* __restrict__ im,
                  const float* __restrict__ alpha, __bf16* __restrict__ w)
{
    __shared__ float red[256];
    long long row = blockIdx.x;
    int t = threadIdx.x;
    long long i = row * 256 + t;
    float rv = re[i], iv = im[i];
    float v = sqrtf(rv * rv + iv * iv) / alpha[0];
    red[t] = v;
    __syncthreads();
    for (int s = 128; s > 0; s >>= 1) {
        if (t < s) red[t] = fmaxf(red[t], red[t + s]);
        __syncthreads();
    }
    float m = red[0];
    __syncthreads();
    float e = __expf(v - m);
    red[t] = e;
    __syncthreads();
    for (int s = 128; s > 0; s >>= 1) {
        if (t < s) red[t] += red[t + s];
        __syncthreads();
    }
    w[i] = (__bf16)(e / red[0]);
}

// elem = FQ .* FK (complex, channel-major in) -> bf16 pixel-major [b][n][d]
__global__ void elem_pack(const float* __restrict__ qr, const float* __restrict__ qi,
                          const float* __restrict__ kr, const float* __restrict__ ki,
                          __bf16* __restrict__ er, __bf16* __restrict__ ei)
{
    long long idx = (long long)blockIdx.x * 256 + threadIdx.x; // [b][n][d]
    int d = (int)(idx & 255);
    long long bn = idx >> 8;
    int n = (int)(bn & 4095);
    int b = (int)(bn >> 12);
    long long s = ((long long)(b * 256 + d)) * HW + n;
    float a = qr[s], bb = qi[s], c = kr[s], dd = ki[s];
    er[idx] = (__bf16)(a * c - bb * dd);
    ei[idx] = (__bf16)(a * dd + bb * c);
}

// output = beta*(Q*cfr + V - lambd*V*cfr) + (1-beta)*Fw   -> bf16 pixel-major
__global__ void final_blend(const float* __restrict__ Q, const float* __restrict__ V,
                            const float* __restrict__ cfr, const float* __restrict__ Fw,
                            const float* __restrict__ lambd, const float* __restrict__ beta,
                            __bf16* __restrict__ outp)
{
    long long idx = (long long)blockIdx.x * 256 + threadIdx.x; // pixel-major
    int c = (int)(idx & 255);
    long long bp = idx >> 8;
    int p = (int)(bp & 4095);
    int b = (int)(bp >> 12);
    long long n = ((long long)(b * 256 + c)) * HW + p;
    float q = Q[n], v = V[n], cf = cfr[n], fw = Fw[idx];
    float l = lambd[0], be = beta[0];
    float fo = q * cf + (v - l * v * cf);
    outp[idx] = (__bf16)(be * fo + (1.f - be) * fw);
}

// ---------------------------------------------------------------------------
// Host orchestration
// ---------------------------------------------------------------------------
extern "C" void kernel_launch(void* const* d_in, const int* in_sizes, int n_in,
                              void* d_out, int out_size, void* d_ws, size_t ws_size,
                              hipStream_t stream)
{
    const float* x1     = (const float*)d_in[0];
    const float* x2     = (const float*)d_in[1];
    const float* wr     = (const float*)d_in[2];
    const float* br     = (const float*)d_in[3];
    const float* wn     = (const float*)d_in[4];
    const float* bn     = (const float*)d_in[5];
    const float* mlp1   = (const float*)d_in[6];
    const float* mlp2   = (const float*)d_in[7];
    const float* pqw    = (const float*)d_in[8];
    const float* pqb    = (const float*)d_in[9];
    const float* dqw    = (const float*)d_in[10];
    const float* dqb    = (const float*)d_in[11];
    const float* pkw    = (const float*)d_in[12];
    const float* pkb    = (const float*)d_in[13];
    const float* dkw    = (const float*)d_in[14];
    const float* dkb    = (const float*)d_in[15];
    const float* pvw    = (const float*)d_in[16];
    const float* pvb    = (const float*)d_in[17];
    const float* dvw    = (const float*)d_in[18];
    const float* dvb    = (const float*)d_in[19];
    const float* wo     = (const float*)d_in[20];
    const float* bo     = (const float*)d_in[21];
    const float* alpha  = (const float*)d_in[22];
    const float* lambd  = (const float*)d_in[23];
    const float* beta   = (const float*)d_in[24];
    float* out = (float*)d_out;

    // workspace carve-up
    char* ws = (char*)d_ws;
    size_t off = 0;
    auto alloc = [&](size_t bytes) {
        void* p = ws + off;
        off = (off + bytes + 255) & ~(size_t)255;
        return p;
    };
    __bf16* X1p   = (__bf16*)alloc((size_t)NPIX * 256 * 2);
    __bf16* X2p   = (__bf16*)alloc((size_t)NPIX * 128 * 2);
    __bf16* wr_b  = (__bf16*)alloc(65536 * 2);
    __bf16* wn_b  = (__bf16*)alloc(32768 * 2);
    __bf16* wq_b  = (__bf16*)alloc(65536 * 2);
    __bf16* wk_b  = (__bf16*)alloc(65536 * 2);
    __bf16* wv_b  = (__bf16*)alloc(65536 * 2);
    __bf16* wo_b  = (__bf16*)alloc(65536 * 2);
    float*  FR    = (float*) alloc((size_t)TOTE * 4);
    float*  FN    = (float*) alloc((size_t)TOTE * 4);
    float*  pooled= (float*) alloc(NB * NC * 4);
    float*  attn2 = (float*) alloc(NB * 2 * NC * 4);
    float*  Fw    = (float*) alloc((size_t)TOTE * 4);
    __bf16* Fwp   = (__bf16*)alloc((size_t)TOTE * 2);
    float*  Qpre  = (float*) alloc((size_t)TOTE * 4);
    float*  Kpre  = (float*) alloc((size_t)TOTE * 4);
    float*  Vpre  = (float*) alloc((size_t)TOTE * 4);
    float*  Qn    = (float*) alloc((size_t)TOTE * 4);
    float*  Kn    = (float*) alloc((size_t)TOTE * 4);
    float*  Vn    = (float*) alloc((size_t)TOTE * 4);
    float*  FQr   = (float*) alloc((size_t)TOTE * 4);
    float*  FQi   = (float*) alloc((size_t)TOTE * 4);
    float*  FKr   = (float*) alloc((size_t)TOTE * 4);
    float*  FKi   = (float*) alloc((size_t)TOTE * 4);
    __bf16* fqr   = (__bf16*)alloc((size_t)TOTE * 2);
    __bf16* fqi   = (__bf16*)alloc((size_t)TOTE * 2);
    __bf16* fkr   = (__bf16*)alloc((size_t)TOTE * 2);
    __bf16* fki   = (__bf16*)alloc((size_t)TOTE * 2);
    float*  aRe   = (float*) alloc((size_t)NB * NC * NC * 4);
    float*  aIm   = (float*) alloc((size_t)NB * NC * NC * 4);
    __bf16* attnw = (__bf16*)alloc((size_t)NB * NC * NC * 2);
    __bf16* er    = (__bf16*)alloc((size_t)TOTE * 2);
    __bf16* ei    = (__bf16*)alloc((size_t)TOTE * 2);
    float*  Fcr   = (float*) alloc((size_t)TOTE * 4);
    float*  Fci   = (float*) alloc((size_t)TOTE * 4);
    float*  cfr   = (float*) alloc((size_t)TOTE * 4);
    __bf16* blendp= (__bf16*)alloc((size_t)TOTE * 2);
    float*  outp  = (float*) alloc((size_t)TOTE * 4);
    (void)ws_size; (void)n_in; (void)in_sizes; (void)out_size;

    const int EW_BLOCKS = (int)(TOTE / 256);     // elementwise grids (c-fastest)
    const long long sAttn = (long long)NC * HW;  // per-batch stride of [C][HW]
    const long long sMat  = (long long)NC * NC;

    // dispatch over (bias, accum) template specializations
    auto gemm = [&](const __bf16* A, const __bf16* B, float* D, const float* bias,
                    int M, int N, int K, int lda, int ldb, int ldd,
                    long long sA, long long sB, long long sD, int nb,
                    float scale, int accum) {
        dim3 g(M / 128, N / 64, nb), blk(128);
        if (bias)
            gemm_abT_bf16<true, false><<<g, blk, 0, stream>>>(
                A, B, D, bias, K, lda, ldb, ldd, sA, sB, sD, scale);
        else if (accum)
            gemm_abT_bf16<false, true><<<g, blk, 0, stream>>>(
                A, B, D, bias, K, lda, ldb, ldd, sA, sB, sD, scale);
        else
            gemm_abT_bf16<false, false><<<g, blk, 0, stream>>>(
                A, B, D, bias, K, lda, ldb, ldd, sA, sB, sD, scale);
    };

    // ---- weight + activation packing --------------------------------------
    pack_flat_bf16<<<(65536 + 255) / 256, 256, 0, stream>>>(wr, wr_b, 65536);
    pack_flat_bf16<<<(32768 + 255) / 256, 256, 0, stream>>>(wn, wn_b, 32768);
    pack_flat_bf16<<<(65536 + 255) / 256, 256, 0, stream>>>(pqw, wq_b, 65536);
    pack_flat_bf16<<<(65536 + 255) / 256, 256, 0, stream>>>(pkw, wk_b, 65536);
    pack_flat_bf16<<<(65536 + 255) / 256, 256, 0, stream>>>(pvw, wv_b, 65536);
    pack_flat_bf16<<<(65536 + 255) / 256, 256, 0, stream>>>(wo, wo_b, 65536);
    pack_nchw_to_pix_bf16<<<EW_BLOCKS, 256, 0, stream>>>(x1, X1p, 256);
    pack_nchw_to_pix_bf16<<<EW_BLOCKS / 2, 256, 0, stream>>>(x2, X2p, 128);

    // ---- FR / FN (conv1x1 GEMMs, pixel-major) -----------------------------
    gemm(X1p, wr_b, FR, br, NPIX, 256, 256, 256, 256, 256, 0, 0, 0, 1, 1.f, 0);
    gemm(X2p, wn_b, FN, bn, NPIX, 256, 128, 128, 128, 256, 0, 0, 0, 1, 1.f, 0);

    // ---- SE gate ----------------------------------------------------------
    pool_mean<<<NB, 256, 0, stream>>>(FR, FN, pooled);
    mlp_gate<<<1, 512, 0, stream>>>(pooled, mlp1, mlp2, attn2);
    blend_fw<<<EW_BLOCKS, 256, 0, stream>>>(FR, FN, attn2, Fw, Fwp);

    // ---- Q/K/V = dw3x3(conv1x1(Fw)) ---------------------------------------
    gemm(Fwp, wq_b, Qpre, pqb, NPIX, 256, 256, 256, 256, 256, 0, 0, 0, 1, 1.f, 0);
    gemm(Fwp, wk_b, Kpre, pkb, NPIX, 256, 256, 256, 256, 256, 0, 0, 0, 1, 1.f, 0);
    gemm(Fwp, wv_b, Vpre, pvb, NPIX, 256, 256, 256, 256, 256, 0, 0, 0, 1, 1.f, 0);
    dwconv3x3<<<EW_BLOCKS, 256, 0, stream>>>(Qpre, dqw, dqb, Qn);
    dwconv3x3<<<EW_BLOCKS, 256, 0, stream>>>(Kpre, dkw, dkb, Kn);
    dwconv3x3<<<EW_BLOCKS, 256, 0, stream>>>(Vpre, dvw, dvb, Vn);

    // ---- forward FFT2 of Q and K (f32, + bf16 copies for WMMA) ------------
    size_t fft_lds = (size_t)(4 * 4096 + 128) * sizeof(float);
    fft2_plane<<<NB * NC, 256, fft_lds, stream>>>(Qn, nullptr, FQr, FQi, fqr, fqi,
                                                  -1.f, 1.f);
    fft2_plane<<<NB * NC, 256, fft_lds, stream>>>(Kn, nullptr, FKr, FKi, fkr, fki,
                                                  -1.f, 1.f);

    // ---- complex Gram matrix: attn = FQ * FK^T (4 bf16 GEMMs, batched) ----
    gemm(fqr, fkr, aRe, nullptr, NC, NC, HW, HW, HW, NC, sAttn, sAttn, sMat, NB,  1.f, 0);
    gemm(fqi, fki, aRe, nullptr, NC, NC, HW, HW, HW, NC, sAttn, sAttn, sMat, NB, -1.f, 1);
    gemm(fqr, fki, aIm, nullptr, NC, NC, HW, HW, HW, NC, sAttn, sAttn, sMat, NB,  1.f, 0);
    gemm(fqi, fkr, aIm, nullptr, NC, NC, HW, HW, HW, NC, sAttn, sAttn, sMat, NB,  1.f, 1);

    // ---- softmax(|attn|/alpha) and elem = FQ.*FK --------------------------
    attn_softmax<<<NB * NC, 256, 0, stream>>>(aRe, aIm, alpha, attnw);
    elem_pack<<<EW_BLOCKS, 256, 0, stream>>>(FQr, FQi, FKr, FKi, er, ei);

    // ---- F_CFR = attn_w @ elem  (real x complex, 2 bf16 GEMMs) ------------
    gemm(attnw, er, Fcr, nullptr, NC, HW, NC, NC, NC, HW, sMat, sAttn, sAttn, NB, 1.f, 0);
    gemm(attnw, ei, Fci, nullptr, NC, HW, NC, NC, NC, HW, sMat, sAttn, sAttn, NB, 1.f, 0);

    // ---- inverse FFT2 -> real cfr -----------------------------------------
    fft2_plane<<<NB * NC, 256, fft_lds, stream>>>(Fcr, Fci, cfr, nullptr, nullptr,
                                                  nullptr, 1.f, 1.f / HW);

    // ---- modulation + final conv1x1 ---------------------------------------
    final_blend<<<EW_BLOCKS, 256, 0, stream>>>(Qn, Vn, cfr, Fw, lambd, beta, blendp);
    gemm(blendp, wo_b, outp, bo, NPIX, 256, 256, 256, 256, 256, 0, 0, 0, 1, 1.f, 0);
    to_nchw<<<EW_BLOCKS, 256, 0, stream>>>(outp, out);
}
